// DLinearTemporal_11476152615030
// MI455X (gfx1250) — matrix operand order repaired and checked by reading, not compiled, not run on hardware
//
#include <hip/hip_runtime.h>

// ---------------------------------------------------------------------------
// DLinear temporal: moving-average decomposition + per-node dual linear heads.
// B=128, T=336, N=325, D=3, O=96, window=25.
//   rows = 124800 (= 325 groups * 384 rows), GEMM per group: (384 x 672)@(672 x 96)
//   (concatenated K: res|mean vs W_season|W_trend), K padded to 704 (22 tiles of 32).
// bf16 WMMA (v_wmma_f32_16x16x32_bf16), fp32 accumulate.
// ---------------------------------------------------------------------------

typedef __bf16  v16bf __attribute__((ext_vector_type(16)));
typedef float   v8f   __attribute__((ext_vector_type(8)));

#define T_LEN   336
#define ND      975          // N*D
#define O_LEN   96
#define KHALF   352          // 336 + 16 pad
#define KPAD    704          // 2*KHALF, 22 k-tiles of 32
#define LROW    712          // LDS A row stride (bf16 elems), 16B aligned rows
#define WIN     25

union AB32 {                 // 32-byte fragment assembled from two b128 loads
    uint4 q[2];
    v16bf v;
};

__device__ __forceinline__ unsigned short f2bf(float f) {
    unsigned int u = __builtin_bit_cast(unsigned int, f);
    unsigned int r = u + 0x7FFFu + ((u >> 16) & 1u);   // round-to-nearest-even
    return (unsigned short)(r >> 16);
}

// ---------------------------------------------------------------------------
// Kernel 1: stage W_season/W_trend -> bf16, transposed to (g, o, k) with K pad.
// One block per group g. LDS tile transpose keeps global R/W coalesced.
// ---------------------------------------------------------------------------
__global__ void __launch_bounds__(256)
stage_w_kernel(const float* __restrict__ Wseason,
               const float* __restrict__ Wtrend,
               unsigned short* __restrict__ Wcat) {
    __shared__ unsigned short st[O_LEN * 340];   // 96 x 336 transposed, padded stride
    const int g = blockIdx.x;

    for (int srci = 0; srci < 2; ++srci) {
        const float* W = (srci == 0 ? Wseason : Wtrend) + (size_t)g * T_LEN * O_LEN;
        const int koff = srci * KHALF;
        if (srci) __syncthreads();               // LDS reuse fence

        // read (t,o) coalesced, write LDS transposed (o,t)
        for (int idx = threadIdx.x; idx < T_LEN * O_LEN; idx += 256) {
            const int t = idx / O_LEN;
            const int o = idx - t * O_LEN;
            st[o * 340 + t] = f2bf(W[idx]);
        }
        __syncthreads();

        // write (o,t) rows coalesced along t
        for (int idx = threadIdx.x; idx < O_LEN * T_LEN; idx += 256) {
            const int o = idx / T_LEN;
            const int t = idx - o * T_LEN;
            Wcat[((size_t)g * O_LEN + o) * KPAD + koff + t] = st[o * 340 + t];
        }
        // zero K pad [336,352)
        for (int idx = threadIdx.x; idx < O_LEN * 16; idx += 256) {
            const int o  = idx >> 4;
            const int kk = idx & 15;
            Wcat[((size_t)g * O_LEN + o) * KPAD + koff + T_LEN + kk] = 0;
        }
    }
}

// ---------------------------------------------------------------------------
// Kernel 2: fused decomposition + GEMM + scrambled epilogue.
// grid = (3 row-blocks, 325 groups); block = 256 threads = 8 waves.
// LDS: A operand, 128 rows x 704 bf16 (stride 712) = 182,272 B.
// ---------------------------------------------------------------------------
extern "C" __global__ void __launch_bounds__(256, 1)
dlinear_fused_kernel(const float* __restrict__ x,
                     const unsigned short* __restrict__ Wcat,
                     const float* __restrict__ bseason,
                     const float* __restrict__ btrend,
                     float* __restrict__ out) {
    extern __shared__ unsigned short sA[];      // [128][LROW]

    const int g   = blockIdx.y;                 // group 0..324
    const int blk = blockIdx.x;                 // row block 0..2
    const int tid = threadIdx.x;

    // ---------------- Phase A: moving average -> res|mean bf16 rows in LDS ---
    {
        const int r = tid & 127;                // LDS row
        const int h = tid >> 7;                 // T half (0: t<168, 1: t>=168)
        const int rflat = g * 384 + blk * 128 + r;
        const int b  = rflat / ND;
        const int nd = rflat - b * ND;
        const float* zp = x + (size_t)b * (T_LEN * ND) + nd;   // elem t at zp[t*ND]

        auto zat = [&](int t) -> float {
            t = t < 0 ? 0 : (t > T_LEN - 1 ? T_LEN - 1 : t);
            return zp[(size_t)t * ND];
        };

        const int t0 = h * (T_LEN / 2);
        float sum = 0.f;
        #pragma unroll
        for (int j = -12; j <= 12; ++j) sum += zat(t0 + j);

        unsigned short* rowp = sA + r * LROW;
        for (int t = t0; t < t0 + T_LEN / 2; ++t) {
            const float mean = sum * (1.0f / (float)WIN);
            const float zt   = zat(t);
            rowp[t]         = f2bf(zt - mean);   // res  -> K [0,336)
            rowp[KHALF + t] = f2bf(mean);        // mean -> K [352,688)
            sum += zat(t + 13) - zat(t - 12);
        }
        if (h == 0) { for (int k = T_LEN; k < KHALF; ++k)         rowp[k] = 0; }
        else        { for (int k = KHALF + T_LEN; k < KPAD; ++k)  rowp[k] = 0; }
    }
    __syncthreads();

    // ---------------- Phase B: WMMA GEMM ------------------------------------
    const int lane = tid & 31;
    const int w    = tid >> 5;        // wave id 0..7 -> M-tile (16 rows each)
    const int lmod = lane & 15;
    const int lhi  = lane >> 4;

    v8f acc[6];
    #pragma unroll
    for (int n = 0; n < 6; ++n) acc[n] = (v8f){0.f, 0.f, 0.f, 0.f, 0.f, 0.f, 0.f, 0.f};

    const unsigned short* wg = Wcat + (size_t)g * O_LEN * KPAD;
    const unsigned short* arow = sA + (w * 16 + lmod) * LROW;

    for (int kt = 0; kt < KPAD / 32; ++kt) {
        const int kb = kt * 32;
        // A fragment (16-bit 16x32 layout): lane holds row M=lane%16,
        // K chunks [kb+8*lhi, +8) and [kb+16+8*lhi, +8)
        AB32 ua;
        const unsigned short* ap = arow + kb + 8 * lhi;
        ua.q[0] = *(const uint4*)(ap);
        ua.q[1] = *(const uint4*)(ap + 16);

        #pragma unroll
        for (int n = 0; n < 6; ++n) {
            // B fragment: col o = n*16+lane%16, K chunks [kb+8*lhi, +8) and [kb+16+8*lhi, +8)
            AB32 ub;
            const unsigned short* bp = wg + (size_t)(n * 16 + lmod) * KPAD + kb + 8 * lhi;
            ub.q[0] = *(const uint4*)(bp);
            ub.q[1] = *(const uint4*)(bp + 16);
            acc[n] = __builtin_amdgcn_wmma_f32_16x16x32_bf16(
                false, ua.v, false, ub.v, (short)0, acc[n], false, false);
        }
    }

    // ---------------- Epilogue: biases + torch-view scramble ----------------
    // C layout: VGPR v holds M = v + 8*(lane>>4), col = lane%16.
    // f = j*31200 + g*96 + o ; view (128,325,3,96) ; out[b][o][n2][d]
    const int jbase = blk * 128 + w * 16 + 8 * lhi;
    #pragma unroll
    for (int n = 0; n < 6; ++n) {
        const int o = n * 16 + lmod;
        const float bias = bseason[g * O_LEN + o] + btrend[g * O_LEN + o];
        #pragma unroll
        for (int v = 0; v < 8; ++v) {
            const int j    = jbase + v;
            const int f    = j * 31200 + g * O_LEN + o;
            const int b    = f / 93600;
            const int rem  = f - b * 93600;
            const int n2   = rem / 288;
            const int rem2 = rem - n2 * 288;
            const int d    = rem2 / O_LEN;       // col within 288 block
            out[(size_t)b * 93600 + (size_t)o * ND + n2 * 3 + d] = acc[n][v] + bias;
        }
    }
}

// ---------------------------------------------------------------------------
extern "C" void kernel_launch(void* const* d_in, const int* in_sizes, int n_in,
                              void* d_out, int out_size, void* d_ws, size_t ws_size,
                              hipStream_t stream) {
    (void)in_sizes; (void)n_in; (void)out_size; (void)ws_size;
    const float* x  = (const float*)d_in[0];
    const float* Ws = (const float*)d_in[1];
    const float* bs = (const float*)d_in[2];
    const float* Wt = (const float*)d_in[3];
    const float* bt = (const float*)d_in[4];
    float* out = (float*)d_out;

    unsigned short* Wcat = (unsigned short*)d_ws;   // 325*96*704*2 B = 43.9 MB

    stage_w_kernel<<<325, 256, 0, stream>>>(Ws, Wt, Wcat);

    dim3 grid(3, 325);
    const size_t ldsA = 128 * LROW * sizeof(unsigned short);  // 182,272 B
    dlinear_fused_kernel<<<grid, 256, ldsA, stream>>>(x, Wcat, bs, bt, out);
}
